// MixtureOfUnits_75282186764335
// MI455X (gfx1250) — matrix operand order repaired and checked by reading, not compiled, not run on hardware
//
#include <hip/hip_runtime.h>
#include <hip/hip_bf16.h>

// ---------------------------------------------------------------------------
// MixtureOfUnits for MI455X (gfx1250, wave32, WMMA).
// Heavy compute = grouped 3x3 conv on the TOP_K=2 selected units per batch,
// done as implicit GEMM with v_wmma_f32_16x16x32_bf16.
// ---------------------------------------------------------------------------

typedef __attribute__((ext_vector_type(16))) __bf16 v16bf;
typedef __attribute__((ext_vector_type(8)))  float  v8f;
typedef __attribute__((ext_vector_type(8)))  unsigned int v8u;

#define Bt 8
#define Ct 128
#define Ht 64
#define Wt 64
#define Ut 8
#define TOPK 2
#define Lt 10
#define HIDt 128
#define NSEL (Bt * TOPK)          // 16 selected (b,u) pairs
#define CH_STRIDE 136             // padded channel stride (halves) for LDS tiles

// output layout (floats), concatenated in reference return order
#define OUT_FINAL 0
#define OUT_CLS   (Bt * Ct * Ht * Wt)          // 4194304
#define OUT_L2    (OUT_CLS + Bt * Lt)          // +80
#define OUT_TOPI  (OUT_L2 + 1)
#define OUT_LB    (OUT_TOPI + Bt * TOPK)

// workspace layout (bytes)
#define WS_WPACK   0                                  // U*9*4*8*32*16 bf16 = 2359296 B
#define WS_CONVOUT (WS_WPACK + 2359296)               // 16*128*64*64 f32 = 33554432 B
#define WS_POOLED  (WS_CONVOUT + 33554432)            // 1024 f32
#define WS_SUMS    (WS_POOLED + 4096)                 // 16 f32
#define WS_SEL     (WS_SUMS + 64)                     // 16 int
#define WS_MASK    (WS_SEL + 64)                      // 64 int
#define WS_GATE    (WS_MASK + 256)                    // 64 f32

__device__ __forceinline__ unsigned short f32_to_bf16_rne(float f) {
  union { float f; unsigned int u; } c; c.f = f;
  unsigned int u = c.u;
  unsigned int r = u + 0x7FFFu + ((u >> 16) & 1u);
  return (unsigned short)(r >> 16);
}

__device__ __forceinline__ v16bf load_frag32B(const unsigned short* p) {
  const uint4* q = (const uint4*)p;     // p is 16B aligned by construction
  uint4 lo = q[0], hi = q[1];
  v8u t;
  t[0] = lo.x; t[1] = lo.y; t[2] = lo.z; t[3] = lo.w;
  t[4] = hi.x; t[5] = hi.y; t[6] = hi.z; t[7] = hi.w;
  return __builtin_bit_cast(v16bf, t);
}

// ---------------------------------------------------------------------------
// 0. zero tiny accumulators
// ---------------------------------------------------------------------------
__global__ void mou_zero_kernel(float* sums) {
  if (threadIdx.x < NSEL) sums[threadIdx.x] = 0.0f;
}

// ---------------------------------------------------------------------------
// 1. adaptive average pool: pooled[b][c] = mean over H*W
// ---------------------------------------------------------------------------
__global__ void mou_pool_kernel(const float* __restrict__ x, float* __restrict__ pooled) {
  int bc = blockIdx.x;                          // 0..1023
  const float* px = x + (size_t)bc * (Ht * Wt);
  float s = 0.0f;
  for (int i = threadIdx.x; i < Ht * Wt; i += 256) s += px[i];
  __shared__ float red[256];
  red[threadIdx.x] = s;
  __syncthreads();
  for (int off = 128; off > 0; off >>= 1) {
    if ((int)threadIdx.x < off) red[threadIdx.x] += red[threadIdx.x + off];
    __syncthreads();
  }
  if (threadIdx.x == 0) pooled[bc] = red[0] * (1.0f / (Ht * Wt));
}

// ---------------------------------------------------------------------------
// 2. router: MLP chain + top-k + losses (one block, 256 threads)
// ---------------------------------------------------------------------------
__device__ void mou_dense(const float* in, const float* Wm, const float* bias,
                          float* out, int ci, int co, bool relu, int tid) {
  for (int t = tid; t < Bt * co; t += 256) {
    int bb = t / co, o = t - bb * co;
    const float* wrow = Wm + (size_t)o * ci;
    const float* iv = in + bb * ci;
    float acc = bias[o];
    for (int k = 0; k < ci; ++k) acc += iv[k] * wrow[k];
    out[t] = relu ? fmaxf(acc, 0.0f) : acc;
  }
}

__global__ void mou_router_kernel(const float* __restrict__ pooled,
                                  const float* W1, const float* b1,
                                  const float* W2, const float* b2,
                                  const float* W3, const float* b3,
                                  const float* W4, const float* b4,
                                  const float* Wu, const float* bu,
                                  const float* Wc, const float* bc,
                                  float* __restrict__ d_out,
                                  int* __restrict__ sel_u,
                                  int* __restrict__ mask) {
  __shared__ float bufA[Bt * HIDt];
  __shared__ float bufB[Bt * HIDt];
  __shared__ float us_s[Bt * Ut];
  __shared__ float cls_s[Bt * Lt];
  __shared__ float red[256];
  __shared__ float l2s;
  int tid = threadIdx.x;

  for (int t = tid; t < Bt * Ct; t += 256) bufA[t] = pooled[t];
  __syncthreads();
  mou_dense(bufA, W1, b1, bufB, 128, 128, true, tid);  __syncthreads();
  mou_dense(bufB, W2, b2, bufA, 128, 128, true, tid);  __syncthreads();
  mou_dense(bufA, W3, b3, bufB, 128, 64, true, tid);   __syncthreads();
  mou_dense(bufB, W4, b4, bufA, 64, 64, false, tid);   __syncthreads();  // feat
  mou_dense(bufA, Wu, bu, us_s, 64, 8, false, tid);
  mou_dense(bufA, Wc, bc, cls_s, 64, 10, false, tid);
  __syncthreads();

  for (int t = tid; t < Bt * Lt; t += 256) d_out[OUT_CLS + t] = cls_s[t];

  // top-k (k=2), first occurrence on ties (matches lax.top_k); noise omitted.
  if (tid < Bt) {
    int b = tid;
    float best = -3.4e38f; int bi = 0;
    for (int u = 0; u < Ut; ++u) {
      float v = us_s[b * Ut + u];
      if (v > best) { best = v; bi = u; }
    }
    float best2 = -3.4e38f; int bi2 = 0;
    for (int u = 0; u < Ut; ++u) {
      if (u == bi) continue;
      float v = us_s[b * Ut + u];
      if (v > best2) { best2 = v; bi2 = u; }
    }
    sel_u[b * TOPK + 0] = bi;
    sel_u[b * TOPK + 1] = bi2;
    for (int u = 0; u < Ut; ++u) mask[b * Ut + u] = (u == bi || u == bi2) ? 1 : 0;
    d_out[OUT_TOPI + b * TOPK + 0] = (float)bi;
    d_out[OUT_TOPI + b * TOPK + 1] = (float)bi2;
  }
  __syncthreads();

  if (tid == 0) {
    float counts[Ut];
    for (int u = 0; u < Ut; ++u) counts[u] = 0.0f;
    for (int s = 0; s < NSEL; ++s) counts[sel_u[s]] += 1.0f;
    float mean = 0.0f;
    for (int u = 0; u < Ut; ++u) { counts[u] *= (1.0f / NSEL); mean += counts[u]; }
    mean *= (1.0f / Ut);
    float lb = 0.0f;
    for (int u = 0; u < Ut; ++u) { float d = counts[u] - mean; lb += d * d; }
    d_out[OUT_LB] = lb * (1.0f / Ut);
    l2s = 0.0f;
  }

  // l2 regularization: 0.01 * sum of Frobenius norms
  const float* ps[12] = { W1, b1, W2, b2, W3, b3, W4, b4, Wu, bu, Wc, bc };
  const int   ns[12] = { 16384, 128, 16384, 128, 8192, 64, 4096, 64, 512, 8, 640, 10 };
  for (int t9 = 0; t9 < 12; ++t9) {
    float sq = 0.0f;
    const float* p = ps[t9];
    for (int i = tid; i < ns[t9]; i += 256) { float v = p[i]; sq += v * v; }
    red[tid] = sq;
    __syncthreads();
    for (int off = 128; off > 0; off >>= 1) {
      if (tid < off) red[tid] += red[tid + off];
      __syncthreads();
    }
    if (tid == 0) l2s += sqrtf(red[0]);
    __syncthreads();
  }
  if (tid == 0) d_out[OUT_L2] = 0.01f * l2s;
}

// ---------------------------------------------------------------------------
// 3. pack conv weights (f32 -> bf16) into per-lane WMMA A-fragment layout:
//    wpack[u][khkw][kt][mt][lane][j], 16 halves (32B) contiguous per lane.
//    A layout used here (16x32 bf16): M = lane%16;
//    half j -> K = (lane>=16 ? 16:0) + j
// ---------------------------------------------------------------------------
__global__ void mou_pack_kernel(const float* __restrict__ conv_w,
                                unsigned short* __restrict__ wpack) {
  int t = blockIdx.x * 256 + threadIdx.x;
  if (t >= Ut * 9 * 4 * 8 * 32 * 16) return;
  int j    = t & 15;
  int lane = (t >> 4) & 31;
  int mt   = (t >> 9) & 7;
  int kt   = (t >> 12) & 3;
  int t2   = t >> 14;
  int khkw = t2 % 9;
  int u    = t2 / 9;
  int m = lane & 15;
  int K = ((lane >= 16) ? 16 : 0) + j;
  int cout = u * Ct + mt * 16 + m;
  int cin  = kt * 32 + K;
  size_t widx = ((size_t)cout * Ct + cin) * 9 + khkw;
  wpack[t] = f32_to_bf16_rne(conv_w[widx]);
}

// ---------------------------------------------------------------------------
// 4. WMMA grouped conv: one block per (selected pair s, output row h).
//    Implicit GEMM: M=128 (Cout), N=64 (W), K=9*128 decomposed as (kh,kw,kt).
// ---------------------------------------------------------------------------
__global__ __launch_bounds__(256)
void mou_conv_kernel(const float* __restrict__ x,
                     const float* __restrict__ conv_b,
                     const unsigned short* __restrict__ wpack,
                     const int* __restrict__ sel_u,
                     float* __restrict__ convout,
                     float* __restrict__ sums) {
  __shared__ unsigned short lds_in[3 * 66 * CH_STRIDE];   // 53856 B
  __shared__ float red[256];

  const int tid  = threadIdx.x;
  const int lane = tid & 31;
  const int wv   = tid >> 5;          // wave id == M tile (0..7)
  const int h    = blockIdx.x;        // output row
  const int s    = blockIdx.y;        // selected pair
  const int u    = sel_u[s];
  const int b    = s >> 1;            // TOPK == 2

  // zero halo columns (input w = -1 and w = 64 -> padded cols 0 and 65)
  for (int idx = tid; idx < 3 * Ct; idx += 256) {
    int r3 = idx >> 7, c = idx & 127;
    lds_in[(r3 * 66 + 0)  * CH_STRIDE + c] = 0;
    lds_in[(r3 * 66 + 65) * CH_STRIDE + c] = 0;
  }

  // stage 3 input rows (h-1, h, h+1) as bf16, [row][w+1][ch] layout
  const float* xb = x + (size_t)b * Ct * Ht * Wt;
  for (int q = tid; q < 3 * Ct * (Wt / 4); q += 256) {   // 6144 float4 loads
    int r3  = q >> 11;
    int rem = q & 2047;
    int c   = rem >> 4;
    int w4  = rem & 15;
    int ir  = h + r3 - 1;
    float4 v = make_float4(0.f, 0.f, 0.f, 0.f);
    if (ir >= 0 && ir < Ht)
      v = *(const float4*)(xb + ((size_t)c * Ht + ir) * Wt + w4 * 4);
    int wb = w4 * 4;
    lds_in[(r3 * 66 + wb + 1) * CH_STRIDE + c] = f32_to_bf16_rne(v.x);
    lds_in[(r3 * 66 + wb + 2) * CH_STRIDE + c] = f32_to_bf16_rne(v.y);
    lds_in[(r3 * 66 + wb + 3) * CH_STRIDE + c] = f32_to_bf16_rne(v.z);
    lds_in[(r3 * 66 + wb + 4) * CH_STRIDE + c] = f32_to_bf16_rne(v.w);
  }
  __syncthreads();

  v8f acc[4];
#pragma unroll
  for (int nt = 0; nt < 4; ++nt)
#pragma unroll
    for (int r = 0; r < 8; ++r) acc[nt][r] = 0.0f;

  const unsigned short* wpu = wpack + (size_t)u * 9 * 4 * 8 * 32 * 16;
  const int n15    = lane & 15;
  const int chhalf = (lane >> 4) << 4;     // 0 or 16: B-frag K offset per lane half

  for (int kh = 0; kh < 3; ++kh) {
#pragma unroll
    for (int kw = 0; kw < 3; ++kw) {
#pragma unroll
      for (int kt = 0; kt < 4; ++kt) {
        const unsigned short* ap =
            wpu + ((((size_t)(kh * 3 + kw) * 4 + kt) * 8 + wv) * 32 + lane) * 16;
        v16bf afrag = load_frag32B(ap);
        const int chbase = kt * 32 + chhalf;
#pragma unroll
        for (int nt = 0; nt < 4; ++nt) {
          // B layout (32x16 bf16): N = lane%16, K = (lane>=16?16:0)+j
          int col = nt * 16 + n15 + kw;            // padded input column
          const unsigned short* bp =
              &lds_in[(kh * 66 + col) * CH_STRIDE + chbase];
          v16bf bfrag = load_frag32B(bp);
          acc[nt] = __builtin_amdgcn_wmma_f32_16x16x32_bf16(
              false, afrag, false, bfrag, (short)0, acc[nt], false, false);
        }
      }
    }
  }

  // epilogue: bias, store, per-unit running sum (for gate mean)
  float lsum = 0.0f;
  const int m_off = (lane >= 16) ? 8 : 0;   // D layout: M = vgpr + 8*(lane>=16)
#pragma unroll
  for (int nt = 0; nt < 4; ++nt) {
    int w = nt * 16 + n15;
#pragma unroll
    for (int r = 0; r < 8; ++r) {
      int cout = wv * 16 + r + m_off;
      float val = acc[nt][r] + conv_b[u * Ct + cout];
      convout[(((size_t)s * Ct + cout) * Ht + h) * Wt + w] = val;
      lsum += val;
    }
  }
  red[tid] = lsum;
  __syncthreads();
  for (int off = 128; off > 0; off >>= 1) {
    if (tid < off) red[tid] += red[tid + off];
    __syncthreads();
  }
  if (tid == 0) atomicAdd(&sums[s], red[0]);
}

// ---------------------------------------------------------------------------
// 5. gate softmax over per-unit means (bias-only mean for unselected units)
// ---------------------------------------------------------------------------
__global__ void mou_gate_kernel(const float* __restrict__ sums,
                                const float* __restrict__ conv_b,
                                const int* __restrict__ sel_u,
                                const int* __restrict__ mask,
                                float* __restrict__ gate) {
  int b = threadIdx.x;
  if (b >= Bt) return;
  float means[Ut];
  for (int u = 0; u < Ut; ++u) {
    if (mask[b * Ut + u]) {
      int s = (sel_u[b * TOPK + 0] == u) ? (b * TOPK + 0) : (b * TOPK + 1);
      means[u] = sums[s] * (1.0f / (Ct * Ht * Wt));
    } else {
      float t = 0.0f;
      for (int c = 0; c < Ct; ++c) t += conv_b[u * Ct + c];
      means[u] = t * (1.0f / Ct);
    }
  }
  float mx = means[0];
  for (int u = 1; u < Ut; ++u) mx = fmaxf(mx, means[u]);
  float den = 0.0f, e[Ut];
  for (int u = 0; u < Ut; ++u) { e[u] = __expf(means[u] - mx); den += e[u]; }
  for (int u = 0; u < Ut; ++u) gate[b * Ut + u] = e[u] / den;
}

// ---------------------------------------------------------------------------
// 6. gated combine: final[b,c,h,w]
// ---------------------------------------------------------------------------
__global__ void mou_combine_kernel(const float* __restrict__ convout,
                                   const float* __restrict__ conv_b,
                                   const float* __restrict__ gate,
                                   const int* __restrict__ sel_u,
                                   const int* __restrict__ mask,
                                   float* __restrict__ d_out) {
  size_t idx = (size_t)blockIdx.x * 256 + threadIdx.x;   // < 8*128*4096
  int p = (int)(idx & 4095);
  int c = (int)((idx >> 12) & 127);
  int b = (int)(idx >> 19);
  float acc = 0.0f;
  for (int k = 0; k < TOPK; ++k) {
    int s = b * TOPK + k;
    acc += gate[b * Ut + sel_u[s]] * convout[((size_t)s * Ct + c) * 4096 + p];
  }
  for (int u = 0; u < Ut; ++u)
    if (!mask[b * Ut + u]) acc += gate[b * Ut + u] * conv_b[u * Ct + c];
  d_out[OUT_FINAL + idx] = acc;
}

// ---------------------------------------------------------------------------
extern "C" void kernel_launch(void* const* d_in, const int* in_sizes, int n_in,
                              void* d_out, int out_size, void* d_ws, size_t ws_size,
                              hipStream_t stream) {
  const float* x      = (const float*)d_in[0];
  const float* W1     = (const float*)d_in[1];
  const float* b1     = (const float*)d_in[2];
  const float* W2     = (const float*)d_in[3];
  const float* b2     = (const float*)d_in[4];
  const float* W3     = (const float*)d_in[5];
  const float* b3     = (const float*)d_in[6];
  const float* W4     = (const float*)d_in[7];
  const float* b4     = (const float*)d_in[8];
  const float* Wu     = (const float*)d_in[9];
  const float* bu     = (const float*)d_in[10];
  const float* Wc     = (const float*)d_in[11];
  const float* bc     = (const float*)d_in[12];
  const float* conv_w = (const float*)d_in[13];
  const float* conv_b = (const float*)d_in[14];
  float* out = (float*)d_out;

  char* ws = (char*)d_ws;
  unsigned short* wpack = (unsigned short*)(ws + WS_WPACK);
  float* convout        = (float*)(ws + WS_CONVOUT);
  float* pooled         = (float*)(ws + WS_POOLED);
  float* sums           = (float*)(ws + WS_SUMS);
  int*   sel_u          = (int*)(ws + WS_SEL);
  int*   mask           = (int*)(ws + WS_MASK);
  float* gate           = (float*)(ws + WS_GATE);

  mou_zero_kernel<<<1, 32, 0, stream>>>(sums);
  mou_pool_kernel<<<Bt * Ct, 256, 0, stream>>>(x, pooled);
  mou_router_kernel<<<1, 256, 0, stream>>>(pooled, W1, b1, W2, b2, W3, b3, W4, b4,
                                           Wu, bu, Wc, bc, out, sel_u, mask);
  mou_pack_kernel<<<(Ut * 9 * 4 * 8 * 32 * 16) / 256, 256, 0, stream>>>(conv_w, wpack);
  dim3 cgrid(Ht, NSEL);
  mou_conv_kernel<<<cgrid, 256, 0, stream>>>(x, conv_b, wpack, sel_u, convout, sums);
  mou_gate_kernel<<<1, 32, 0, stream>>>(sums, conv_b, sel_u, mask, gate);
  mou_combine_kernel<<<(Bt * Ct * Ht * Wt) / 256, 256, 0, stream>>>(
      convout, conv_b, gate, sel_u, mask, out);
}
